// AnaphoricityScorer_9354438770987
// MI455X (gfx1250) — matrix operand order repaired and checked by reading, not compiled, not run on hardware
//
#include <hip/hip_runtime.h>

// ---------------------------------------------------------------------------
// Fused anaphoricity scorer for MI455X (gfx1250, wave32, WMMA).
//   pair = [m | g | m*g | pw]  (m segment factored out: only 2 distinct rows
//   per workgroup -> precompute MH = m @ W1[:, :1024]^T once, 28% fewer WMMAs)
//   H = leakyrelu(pair @ W1^T + b1);  score = rough + H @ Wout^T + bout
// ---------------------------------------------------------------------------

typedef __attribute__((ext_vector_type(16))) __bf16 v16bf;
typedef __attribute__((ext_vector_type(8)))  float  v8f;

#define EMB   1024
#define INF   3136          // 3*EMB + 64
#define HID   1024
#define BN    64            // hidden tile
#define KC    64            // staged K chunk (2 WMMA K-steps)
#define NT    (HID / BN)    // 16
#define INF2  2112          // main-loop K: g(1024) + m*g(1024) + pw(64)
#define NKC   (INF2 / KC)   // 33

// LDS layout (bf16 element offsets); rows padded +8 elems (16B) for
// conflict-free ds_load_b128 (consecutive rows 4 banks apart).
#define GS    1032                    // G row stride
#define PWS   72                      // pw row stride
#define BS    72                      // B (W1 tile) row stride
#define BUFS  (64 * BS)               // one B buffer
#define OFF_M  0                      // 3 x 1024 (row 2 = zeros for MH pad)
#define OFF_G  3072                   // 128 x GS
#define OFF_PW (OFF_G + 128 * GS)     // 128 x PWS
#define OFF_B  (OFF_PW + 128 * PWS)   // 2 x BUFS
#define BF_ELEMS   (OFF_B + 2 * BUFS)
#define OFF_MH     (BF_ELEMS * 2)                 // bytes: 2048 floats
#define OFF_SC     (OFF_MH + 2048 * 4)            // bytes: 128 floats
#define OFF_IDX    (OFF_SC + 128 * 4)             // bytes: 128 ints
#define SMEM_BYTES (OFF_IDX + 128 * 4)

// one-op packed f32x2 -> bf16x2 (RTNE)
static __device__ __forceinline__ unsigned int pk2(float a, float b) {
  unsigned int r;
  asm("v_cvt_pk_bf16_f32 %0, %1, %2" : "=v"(r) : "v"(a), "v"(b));
  return r;
}

// convert+store 4 f32 -> 4 bf16 (one 8B LDS store)
static __device__ __forceinline__ void st4(__bf16* d, float4 f) {
  union { unsigned int w[2]; uint2 u; } s;
  s.w[0] = pk2(f.x, f.y);
  s.w[1] = pk2(f.z, f.w);
  *(uint2*)d = s.u;
}

// two ds_load_b128: elements p[0..7] and p[16..23] -> one 16-elem fragment
static __device__ __forceinline__ v16bf ld16(const __bf16* p) {
  union { v16bf v; uint4 q[2]; } u;
  u.q[0] = *(const uint4*)p;
  u.q[1] = *(const uint4*)(p + 16);
  return u.v;
}

// f32 p[0..7] and p[16..23] from global -> v16bf fragment (B layout: table K order)
static __device__ __forceinline__ v16bf cvt16(const float* p) {
  union { v16bf v; unsigned int w[8]; } u;
#pragma unroll
  for (int q = 0; q < 4; ++q) {
    float4 f = ((const float4*)p)[q + (q >> 1) * 2];
    u.w[2 * q]     = pk2(f.x, f.y);
    u.w[2 * q + 1] = pk2(f.z, f.w);
  }
  return u.v;
}

__global__ __launch_bounds__(256)
__attribute__((amdgpu_waves_per_eu(2)))        // budget regs for 2 waves/SIMD32
void coref_fused(
    const float* __restrict__ all_mentions,   // [10000,1024]
    const float* __restrict__ mentions_batch, // [1024,1024]
    const float* __restrict__ pw_batch,       // [1024,64,64]
    const int*   __restrict__ top_idx,        // [1024,64]
    const float* __restrict__ rough,          // [1024,64]
    const float* __restrict__ W1,             // [1024,3136]
    const float* __restrict__ b1,             // [1024]
    const float* __restrict__ Wout,           // [1024]
    const float* __restrict__ bout,           // [1]
    float* __restrict__ out)                  // [1024,65]
{
  extern __shared__ char smem[];
  __bf16* lds     = (__bf16*)smem;
  __bf16* m_s     = lds + OFF_M;
  __bf16* G_s     = lds + OFF_G;
  __bf16* PW_s    = lds + OFF_PW;
  __bf16* B_s     = lds + OFF_B;
  float*  MH_s    = (float*)(smem + OFF_MH);   // [2][1024]
  float*  score_s = (float*)(smem + OFF_SC);
  int*    idx_s   = (int*)(smem + OFF_IDX);

  const int tid = threadIdx.x;
  const int b0  = blockIdx.x;                 // batch rows 2*b0, 2*b0+1

  if (tid < 128) {
    idx_s[tid]   = top_idx[b0 * 128 + tid];
    score_s[tid] = 0.f;
  }
  __syncthreads();

  // ---- stage pair-feature operands in LDS (f32 -> bf16) ----
  for (int e = tid; e < 3 * EMB / 4; e += 256) {              // mentions + zero row
    int r = e >> 8, c = (e & 255) << 2;
    float4 f = make_float4(0.f, 0.f, 0.f, 0.f);
    if (r < 2) f = *(const float4*)(mentions_batch + (size_t)(2 * b0 + r) * EMB + c);
    st4(m_s + r * EMB + c, f);
  }
  for (int e = tid; e < 128 * EMB / 4; e += 256) {            // gather antecedents
    int r = e >> 8, c = (e & 255) << 2;
    float4 f = *(const float4*)(all_mentions + (size_t)idx_s[r] * EMB + c);
    st4(G_s + r * GS + c, f);
  }
  for (int e = tid; e < 128 * 64 / 4; e += 256) {             // pairwise feats
    int r = e >> 4, c = (e & 15) << 2;
    float4 f = *(const float4*)(pw_batch + ((size_t)b0 * 128 + r) * 64 + c);
    st4(PW_s + r * PWS + c, f);
  }
  __syncthreads();

  const int lane   = tid & 31;
  const int wv     = tid >> 5;
  const int jn     = wv & 3;                  // N subtile 0..3
  const int mtop   = wv >> 2;                 // 0/1 -> M subtiles {mtop+2t}
  const int l15    = lane & 15;
  const int hi     = lane & 16;
  const int koff   = hi ? 8 : 0;              // ISA 16-bit A layout K offset
  const int nlocal = jn * 16 + l15;

  // ---- MH = m @ W1[:, :1024]^T : M padded 2->16, one WMMA pass ----
  // Keep rolled: one-time phase (~5% of WMMAs); full unroll blows VGPR budget.
  {
    const __bf16* arow = m_s + ((l15 < 2) ? l15 : 2) * EMB;   // row 2 = zeros
#pragma unroll 1
    for (int q = 0; q < 8; ++q) {
      const int n2 = (wv * 8 + q) * 16 + l15;                 // hidden column
      const float* wp = W1 + (size_t)n2 * INF + (hi ? 8 : 0);
      v8f acc = (v8f){0.f,0.f,0.f,0.f,0.f,0.f,0.f,0.f};
#pragma unroll 2
      for (int k = 0; k < EMB; k += 32) {
        v16bf bf = cvt16(wp + k);
        v16bf af = ld16(arow + k + koff);
        acc = __builtin_amdgcn_wmma_f32_16x16x32_bf16(
            false, af, false, bf, (short)0, acc, false, false);
      }
      if (!hi) { MH_s[n2] = acc[0]; MH_s[EMB + n2] = acc[1]; }
    }
  }
  __syncthreads();

  float sacc[4][8];
#pragma unroll
  for (int t = 0; t < 4; ++t)
#pragma unroll
    for (int v = 0; v < 8; ++v) sacc[t][v] = 0.f;

  for (int nt = 0; nt < NT; ++nt) {
    const int n0 = nt * BN;
    v8f acc[4];
#pragma unroll
    for (int t = 0; t < 4; ++t) acc[t] = (v8f){0.f,0.f,0.f,0.f,0.f,0.f,0.f,0.f};

    for (int kc = 0; kc < NKC; ++kc) {
      { // stage 64x64 W1 chunk (cols 1024 + kc*64 ..) -> bf16, double buffered
        const int n = tid >> 2, part = tid & 3;
        const float* src = W1 + (size_t)(n0 + n) * INF + EMB + kc * KC + part * 16;
        __bf16* dst = B_s + (kc & 1) * BUFS + n * BS + part * 16;
#pragma unroll
        for (int q = 0; q < 4; ++q) st4(dst + q * 4, ((const float4*)src)[q]);
      }
      __syncthreads();

      const __bf16* bb = B_s + (kc & 1) * BUFS + nlocal * BS;
#pragma unroll
      for (int ks = 0; ks < 2; ++ks) {
        const int kg = kc * KC + ks * 32;     // uniform -> scalar branches
        v16bf bfrag = ld16(bb + ks * 32 + (hi ? 8 : 0));
        if (kg < EMB) {                                 // segment: g (gathered)
#pragma unroll
          for (int t = 0; t < 4; ++t) {
            const int row = (mtop + 2 * t) * 16 + l15;
            v16bf a = ld16(G_s + row * GS + kg + koff);
            acc[t] = __builtin_amdgcn_wmma_f32_16x16x32_bf16(
                false, a, false, bfrag, (short)0, acc[t], false, false);
          }
        } else if (kg < 2 * EMB) {                      // segment: m*g on the fly
          const int k2 = kg - EMB;
          union U { v16bf v; unsigned int w[8]; };
          U mm0, mm1;
          mm0.v = ld16(m_s + k2 + koff);                // bi = 0
          mm1.v = ld16(m_s + EMB + k2 + koff);          // bi = 1
#pragma unroll
          for (int t = 0; t < 4; ++t) {
            const int row = (mtop + 2 * t) * 16 + l15;
            U g, r;
            g.v = ld16(G_s + row * GS + k2 + koff);
            const U& mm = (t < 2) ? mm0 : mm1;
#pragma unroll
            for (int qq = 0; qq < 8; ++qq)
              asm("v_pk_mul_bf16 %0, %1, %2"
                  : "=v"(r.w[qq]) : "v"(g.w[qq]), "v"(mm.w[qq]));
            acc[t] = __builtin_amdgcn_wmma_f32_16x16x32_bf16(
                false, r.v, false, bfrag, (short)0, acc[t], false, false);
          }
        } else {                                        // segment: pw
          const int k3 = kg - 2 * EMB;
#pragma unroll
          for (int t = 0; t < 4; ++t) {
            const int row = (mtop + 2 * t) * 16 + l15;
            v16bf a = ld16(PW_s + row * PWS + k3 + koff);
            acc[t] = __builtin_amdgcn_wmma_f32_16x16x32_bf16(
                false, a, false, bfrag, (short)0, acc[t], false, false);
          }
        }
      }
    }

    // epilogue: + MH + bias, leaky relu, fold Wout into per-row partial score
    const int ng = n0 + nlocal;
    const float mh0 = MH_s[ng];
    const float mh1 = MH_s[EMB + ng];
    const float b1v = b1[ng];
    const float wov = Wout[ng];
#pragma unroll
    for (int t = 0; t < 4; ++t) {
      const float base = ((t < 2) ? mh0 : mh1) + b1v;
#pragma unroll
      for (int v = 0; v < 8; ++v) {
        float h = acc[t][v] + base;
        h = (h > 0.f) ? h : 0.01f * h;
        sacc[t][v] += h * wov;
      }
    }
    __syncthreads();   // B_s buffer reuse across N tiles
  }

  // cross-wave / cross-lane reduction over hidden dim
#pragma unroll
  for (int t = 0; t < 4; ++t)
#pragma unroll
    for (int v = 0; v < 8; ++v) {
      const int row = (mtop + 2 * t) * 16 + v + (hi ? 8 : 0);
      atomicAdd(&score_s[row], sacc[t][v]);
    }
  __syncthreads();

  const float bo = bout[0];
  for (int r = tid; r < 128; r += 256) {
    const int bi = 2 * b0 + (r >> 6);
    const int an = r & 63;
    out[(size_t)bi * 65 + 1 + an] = rough[(size_t)b0 * 128 + r] + score_s[r] + bo;
  }
  if (tid < 2) out[(size_t)(2 * b0 + tid) * 65] = 1e-7f;   // dummy eps column
}

extern "C" void kernel_launch(void* const* d_in, const int* in_sizes, int n_in,
                              void* d_out, int out_size, void* d_ws, size_t ws_size,
                              hipStream_t stream) {
  (void)in_sizes; (void)n_in; (void)d_ws; (void)ws_size; (void)out_size;
  const float* all_mentions = (const float*)d_in[0];
  const float* mentions     = (const float*)d_in[1];
  const float* pw           = (const float*)d_in[2];
  const int*   idx          = (const int*)d_in[3];
  const float* rough        = (const float*)d_in[4];
  const float* W1           = (const float*)d_in[5];
  const float* b1           = (const float*)d_in[6];
  const float* Wout         = (const float*)d_in[7];
  const float* bout         = (const float*)d_in[8];
  float* out = (float*)d_out;

  (void)hipFuncSetAttribute((const void*)coref_fused,
                            hipFuncAttributeMaxDynamicSharedMemorySize,
                            SMEM_BYTES);
  coref_fused<<<dim3(512), dim3(256), SMEM_BYTES, stream>>>(
      all_mentions, mentions, pw, idx, rough, W1, b1, Wout, bout, out);
}
